// ICLAttention_14654428413952
// MI455X (gfx1250) — compile-verified, matrix operand order checked by reading, not run on hardware
//
#include <hip/hip_runtime.h>
#include <hip/hip_bf16.h>

// ---------------------------------------------------------------------------
// ICLAttention for MI455X (gfx1250, wave32, WMMA + async-to-LDS)
//
// Pipeline (all matmuls on v_wmma_f32_16x16x32_f16, f32 accumulation):
//   1. qh  = q @ W_q            -> f16 [B][H][S][D]      (gemm, MODE_QK)
//   2. kh  = k @ W_k            -> f16 [B][H][S][D]      (gemm, MODE_QK)
//   3. vt  = (v @ W_v + b_v)^T  -> f16 [B][H][D][S]      (gemm, MODE_V)
//   4. at  = flash_attention(qh, kh, vt) -> f16 [B*S][HID]
//      (K / V^T tiles staged with global_load_async_to_lds_b128, ASYNCcnt)
//   5. out = at @ W_o + b_o     -> f32 d_out             (gemm, MODE_OUT)
// ---------------------------------------------------------------------------

typedef __attribute__((ext_vector_type(16))) _Float16 v16h;
typedef __attribute__((ext_vector_type(8)))  _Float16 v8h;
typedef __attribute__((ext_vector_type(8)))  float    v8f;

#define DEVINL __device__ __forceinline__

// Fragment layout (ISA 7.12.2, 16-bit 16x32 A / 32x16 B, wave32):
//   lane l<16  : row/col = l,    K halves {k0..k0+7} U {k0+16..k0+23}, k0=0
//   lane l>=16 : row/col = l-16, same runs with k0=8
// `p` points at (row, kbase + (lane>=16 ? 8 : 0)) in a row-major f16 buffer.
DEVINL v16h frag_ab(const _Float16* p) {
  v8h lo = *(const v8h*)p;
  v8h hi = *(const v8h*)(p + 16);
  v16h r;
#pragma unroll
  for (int i = 0; i < 8; ++i) { r[i] = lo[i]; r[8 + i] = hi[i]; }
  return r;
}

DEVINL v8f wmma_f16(v16h a, v16h b, v8f c) {
  return __builtin_amdgcn_wmma_f32_16x16x32_f16(false, a, false, b,
                                                (short)0, c, false, false);
}

// Async memory->LDS copy of 32 contiguous bytes per lane (2 x B128).
// GVS mode: addr = SGPR64 base + 32-bit per-lane byte offset + INST_OFFSET,
// INST_OFFSET is added on both the LDS and global side (ISA §10, async ops).
// LDS byte offset = low 32 bits of the generic __shared__ pointer.
DEVINL void async_copy32B(const void* gbase, unsigned goff, void* lds_dst) {
  unsigned loff = (unsigned)(uintptr_t)lds_dst;
  asm volatile(
      "global_load_async_to_lds_b128 %0, %1, %2 offset:0\n\t"
      "global_load_async_to_lds_b128 %0, %1, %2 offset:16"
      :
      : "v"(loff), "v"(goff), "s"((unsigned long long)(uintptr_t)gbase)
      : "memory");
}

DEVINL void wait_async() {
  asm volatile("s_wait_asynccnt 0x0" ::: "memory");
}

// ------------------------------- constants ---------------------------------
constexpr int KDIM = 1024;   // GEMM reduction dim
constexpr int NDIM = 1024;   // GEMM output dim (== HID)
constexpr int SEQ  = 2048;
constexpr int NH   = 16;
constexpr int HD   = 64;

constexpr int BM = 256, BN = 64, BK = 32;
constexpr int LDA = BK + 8;          // LDS row strides (halves), 16B-aligned
constexpr int LDB = BK + 8;

enum { MODE_QK = 0, MODE_V = 1, MODE_OUT = 2 };

// ------------------------------ GEMM kernel --------------------------------
// C[M][N=1024] = A[M][1024] @ W[1024][1024] (+bias), epilogue per MODE.
// 8 waves; each wave computes a 32x64 sub-tile (2 A-frags x 4 B-frags
// -> 8 independent WMMAs per K-step, 1.5 ds_load_b128 per WMMA).
template <int MODE, typename AT>
__global__ __launch_bounds__(256) void gemm_kernel(
    const AT* __restrict__ A, const float* __restrict__ W,
    const float* __restrict__ bias, void* __restrict__ Out) {
  __shared__ _Float16 As[BM * LDA];       // [m][k]
  __shared__ _Float16 Bs[BN * LDB];       // transposed: [n][k]

  const int tid  = threadIdx.x;
  const int lane = tid & 31, wv = tid >> 5;
  const int lh   = lane & 15, hb = lane >> 4;       // half-wave select
  const int m0   = blockIdx.x * BM;
  const int n0   = blockIdx.y * BN;

  v8f acc[2][4];
#pragma unroll
  for (int h = 0; h < 2; ++h)
#pragma unroll
    for (int i = 0; i < 4; ++i)
#pragma unroll
      for (int j = 0; j < 8; ++j) acc[h][i][j] = 0.0f;

  const int bk = tid >> 3;                // 0..31
  const int bn = (tid & 7) * 8;           // 0..56

  for (int k0 = 0; k0 < KDIM; k0 += BK) {
    {   // stage A tile: one full 32-wide row per thread (convert to f16)
      const AT* src = A + (size_t)(m0 + tid) * KDIM + k0;
      _Float16* dst = &As[tid * LDA];
#pragma unroll
      for (int i = 0; i < 32; ++i) dst[i] = (_Float16)src[i];
    }
    {   // stage W tile transposed [n][k]
      const float* src = W + (size_t)(k0 + bk) * NDIM + n0 + bn;
#pragma unroll
      for (int i = 0; i < 8; ++i) Bs[(bn + i) * LDB + bk] = (_Float16)src[i];
    }
    __syncthreads();

    v16h af0 = frag_ab(&As[(wv * 32 + lh) * LDA + hb * 8]);
    v16h af1 = frag_ab(&As[(wv * 32 + 16 + lh) * LDA + hb * 8]);
#pragma unroll
    for (int nt = 0; nt < 4; ++nt) {
      v16h bf = frag_ab(&Bs[(nt * 16 + lh) * LDB + hb * 8]);
      acc[0][nt] = wmma_f16(af0, bf, acc[0][nt]);
      acc[1][nt] = wmma_f16(af1, bf, acc[1][nt]);
    }
    __syncthreads();
  }

  // epilogue: element (row = m0+wv*32+hf*16+hb*8+r, col = n0+nt*16+lh)
#pragma unroll
  for (int hf = 0; hf < 2; ++hf) {
#pragma unroll
    for (int nt = 0; nt < 4; ++nt) {
      const int col = n0 + nt * 16 + lh;
#pragma unroll
      for (int r = 0; r < 8; ++r) {
        const int row = m0 + wv * 32 + hf * 16 + hb * 8 + r;
        float val = acc[hf][nt][r];
        if constexpr (MODE == MODE_QK) {
          const int b = row >> 11, s = row & (SEQ - 1);
          const int h = col >> 6,  d = col & (HD - 1);
          ((_Float16*)Out)[((size_t)(b * NH + h) * SEQ + s) * HD + d] =
              (_Float16)val;
        } else if constexpr (MODE == MODE_V) {
          val += bias[col];
          const int b = row >> 11, s = row & (SEQ - 1);
          const int h = col >> 6,  d = col & (HD - 1);
          ((_Float16*)Out)[((size_t)(b * NH + h) * HD + d) * SEQ + s] =
              (_Float16)val;
        } else {
          val += bias[col];
          ((float*)Out)[(size_t)row * NDIM + col] = val;
        }
      }
    }
  }
}

// --------------------------- flash attention -------------------------------
constexpr int QB  = 128;                 // queries per block (8 waves x 16)
constexpr int KBL = 64;                  // keys per key-block
constexpr int LDK = HD  + 8;             // Kt row stride (halves)
constexpr int LDV = KBL + 8;             // Vt row stride
constexpr int LDP = KBL + 8;             // P  row stride

__global__ __launch_bounds__(256) void attn_kernel(
    const _Float16* __restrict__ qh, const _Float16* __restrict__ kh,
    const _Float16* __restrict__ vt, _Float16* __restrict__ out) {
  __shared__ _Float16 Kt[KBL * LDK];          // [key][d]   (B-side of Q@K^T)
  __shared__ _Float16 Vt[HD * LDV];           // [d][key]   (B-side of P@V)
  __shared__ _Float16 Ps[8 * 16 * LDP];       // per-wave probs [q][key]

  const int tid  = threadIdx.x;
  const int lane = tid & 31, wv = tid >> 5;
  const int lh   = lane & 15, hb = lane >> 4;
  const int bh   = blockIdx.x;                // b*NH + h
  const int qblk = blockIdx.y;
  const int q0   = qblk * QB + wv * 16;       // wave's first query

  // Q fragments held in registers for the whole kernel (d split into 2 k-steps)
  v16h qa[2];
  {
    const _Float16* qp = qh + ((size_t)bh * SEQ + q0 + lh) * HD;
#pragma unroll
    for (int kt = 0; kt < 2; ++kt) qa[kt] = frag_ab(qp + kt * 32 + hb * 8);
  }

  v8f o[4];
  float rm[8], rs[8];
#pragma unroll
  for (int i = 0; i < 4; ++i)
#pragma unroll
    for (int j = 0; j < 8; ++j) o[i][j] = 0.0f;
#pragma unroll
  for (int r = 0; r < 8; ++r) { rm[r] = -__builtin_inff(); rs[r] = 0.0f; }

  const int nkb     = 2 * (qblk + 1);         // causal: keys j <= max i - 1
  const int kb_need = (q0 + 14) >> 6;         // last block with any live key
  const float scale = 0.125f;                 // 1/sqrt(64)

  // per-thread tile-staging assignment: 64 rows x 64 halves, 32B per thread
  const int srow = tid >> 2;                  // 0..63
  const int sseg = (tid & 3) * 16;            // 0/16/32/48 (halves)

  for (int kb = 0; kb < nkb; ++kb) {
    const int j0 = kb * KBL;
    __syncthreads();                          // protect Kt/Vt from prior reads
    // K tile [key][d] and V^T tile [d][key], async DMA into LDS
    async_copy32B(kh, (unsigned)((((unsigned)bh * SEQ + j0 + srow) * HD + sseg) * 2),
                  &Kt[srow * LDK + sseg]);
    async_copy32B(vt, (unsigned)((((unsigned)bh * HD + srow) * SEQ + j0 + sseg) * 2),
                  &Vt[srow * LDV + sseg]);
    wait_async();
    __syncthreads();

    if (kb <= kb_need) {
      // ---- scores = Q @ K^T ----
      v8f sc[4];
#pragma unroll
      for (int nt = 0; nt < 4; ++nt) {
#pragma unroll
        for (int j = 0; j < 8; ++j) sc[nt][j] = 0.0f;
#pragma unroll
        for (int kt = 0; kt < 2; ++kt) {
          v16h bf = frag_ab(&Kt[(nt * 16 + lh) * LDK + kt * 32 + hb * 8]);
          sc[nt] = wmma_f16(qa[kt], bf, sc[nt]);
        }
      }
      // ---- scale + causal mask + row max ----
      float nm[8];
#pragma unroll
      for (int r = 0; r < 8; ++r) nm[r] = rm[r];
#pragma unroll
      for (int nt = 0; nt < 4; ++nt) {
        const int j = j0 + nt * 16 + lh;
#pragma unroll
        for (int r = 0; r < 8; ++r) {
          const int i = q0 + hb * 8 + r;
          float vsc = sc[nt][r] * scale;
          const bool keep = (j < i) || (i == 0 && j == 0);
          vsc = keep ? vsc : -__builtin_inff();
          sc[nt][r] = vsc;
          nm[r] = fmaxf(nm[r], vsc);
        }
      }
#pragma unroll
      for (int r = 0; r < 8; ++r) {       // reduce over the 16 lanes of a row
#pragma unroll
        for (int off = 8; off > 0; off >>= 1)
          nm[r] = fmaxf(nm[r], __shfl_xor(nm[r], off));
      }
      // ---- rescale running state ----
#pragma unroll
      for (int r = 0; r < 8; ++r) {
        const float al = __expf(rm[r] - nm[r]);
        rs[r] *= al;
        rm[r] = nm[r];
#pragma unroll
        for (int nt = 0; nt < 4; ++nt) o[nt][r] *= al;
      }
      // ---- probs, row sums, spill P (C-layout -> A-layout via LDS) ----
      float ps[8];
#pragma unroll
      for (int r = 0; r < 8; ++r) ps[r] = 0.0f;
#pragma unroll
      for (int nt = 0; nt < 4; ++nt) {
#pragma unroll
        for (int r = 0; r < 8; ++r) {
          const float p = __expf(sc[nt][r] - rm[r]);   // -inf -> 0
          ps[r] += p;
          Ps[(wv * 16 + hb * 8 + r) * LDP + nt * 16 + lh] = (_Float16)p;
        }
      }
#pragma unroll
      for (int r = 0; r < 8; ++r) {
#pragma unroll
        for (int off = 8; off > 0; off >>= 1) ps[r] += __shfl_xor(ps[r], off);
        rs[r] += ps[r];
      }
      __syncthreads();
      // ---- O += P @ V ----
#pragma unroll
      for (int kt = 0; kt < 2; ++kt) {
        v16h pf = frag_ab(&Ps[(wv * 16 + lh) * LDP + kt * 32 + hb * 8]);
#pragma unroll
        for (int dt = 0; dt < 4; ++dt) {
          v16h vf = frag_ab(&Vt[(dt * 16 + lh) * LDV + kt * 32 + hb * 8]);
          o[dt] = wmma_f16(pf, vf, o[dt]);
        }
      }
    } else {
      __syncthreads();                    // keep barrier sequence uniform
    }
  }

  // ---- finalize: out[b*S+s][h*64+d] = o / rowsum ----
  const int b = bh >> 4, h = bh & (NH - 1);
#pragma unroll
  for (int nt = 0; nt < 4; ++nt) {
    const int col = h * HD + nt * 16 + lh;
#pragma unroll
    for (int r = 0; r < 8; ++r) {
      const int s = q0 + hb * 8 + r;
      out[((size_t)(b * SEQ + s)) * NDIM + col] = (_Float16)(o[nt][r] / rs[r]);
    }
  }
}

// ------------------------------- launcher ----------------------------------
extern "C" void kernel_launch(void* const* d_in, const int* in_sizes, int n_in,
                              void* d_out, int out_size, void* d_ws,
                              size_t ws_size, hipStream_t stream) {
  (void)in_sizes; (void)n_in; (void)out_size; (void)ws_size;
  const float* q  = (const float*)d_in[0];
  const float* k  = (const float*)d_in[1];
  const float* v  = (const float*)d_in[2];
  const float* Wq = (const float*)d_in[3];
  const float* Wk = (const float*)d_in[4];
  const float* Wv = (const float*)d_in[5];
  const float* bv = (const float*)d_in[6];
  const float* Wo = (const float*)d_in[7];
  const float* bo = (const float*)d_in[8];
  float* out = (float*)d_out;

  const size_t NE = (size_t)2 * SEQ * NDIM;     // 4M f16 elements each
  _Float16* qh = (_Float16*)d_ws;
  _Float16* kh = qh + NE;
  _Float16* vt = kh + NE;
  _Float16* at = vt + NE;

  dim3 grid(4096 / BM, NDIM / BN), blk(256);
  gemm_kernel<MODE_QK, float><<<grid, blk, 0, stream>>>(q, Wq, nullptr, qh);
  gemm_kernel<MODE_QK, float><<<grid, blk, 0, stream>>>(k, Wk, nullptr, kh);
  gemm_kernel<MODE_V,  float><<<grid, blk, 0, stream>>>(v, Wv, bv, vt);
  attn_kernel<<<dim3(2 * NH, SEQ / QB), blk, 0, stream>>>(qh, kh, vt, at);
  gemm_kernel<MODE_OUT, _Float16><<<grid, blk, 0, stream>>>(at, Wo, bo, out);
}
